// Mamba3LayerOfficial_73306501808622
// MI455X (gfx1250) — matrix operand order repaired and checked by reading, not compiled, not run on hardware
//
#include <hip/hip_runtime.h>
#include <hip/hip_bf16.h>

// ---------------------------------------------------------------------------
// Mamba3 layer for gfx1250 (MI455X): bf16 WMMA GEMMs + register-resident scan
// ---------------------------------------------------------------------------

#define D_MODEL   2048
#define D_STATE   64
#define D_INNER   4096
#define N_HEADS   64
#define HEAD_DIM  64
#define BATCH     2
#define SEQ       2048
#define D_PROJ    8384              // 2*D_INNER + 2*D_STATE + N_HEADS
#define SSD_SCALE 0.125f            // (D_STATE*MIMO_RANK)^-0.5
#define EPS       1e-6f

typedef __attribute__((ext_vector_type(16))) __bf16 v16bf;
typedef __attribute__((ext_vector_type(8)))  float  v8f;
typedef __attribute__((ext_vector_type(4)))  int    v4i;
typedef __attribute__((address_space(3))) v4i* lds_v4i_ptr;

// Async global->LDS path (CDNA5): guarded so either toolchain compiles.
#if defined(__HIP_DEVICE_COMPILE__) && \
    __has_builtin(__builtin_amdgcn_global_load_async_to_lds_b128) && \
    __has_builtin(__builtin_amdgcn_s_wait_asynccnt)
#define USE_ASYNC_LDS 1
#else
#define USE_ASYNC_LDS 0
#endif

// ---------------------------------------------------------------------------
// f32 -> bf16 (round-to-nearest-even), stored as raw u16
// ---------------------------------------------------------------------------
__global__ __launch_bounds__(256) void f32_to_bf16_kernel(const float* __restrict__ in,
                                                          unsigned short* __restrict__ out,
                                                          size_t n) {
  size_t i = (size_t)blockIdx.x * blockDim.x + threadIdx.x;
  size_t stride = (size_t)gridDim.x * blockDim.x;
  for (; i < n; i += stride) {
    unsigned int x = __float_as_uint(in[i]);
    unsigned int r = x + 0x7fffu + ((x >> 16) & 1u);
    out[i] = (unsigned short)(r >> 16);
  }
}

// ---------------------------------------------------------------------------
// bf16 GEMM via v_wmma_f32_16x16x32_bf16.
// Block = 128 threads (4 waves). Block tile = 256x64 of C; each wave owns a
// 64x64 quadrant = 16 WMMA accumulators. K stepped by 32 through LDS.
// B fragments (4) are loaded once per K-step and reused against 4 A fragments
// -> 16 WMMAs per 64 fragment dwords per lane.
// A row-major (M x K), B row-major (K x N), C row-major f32 (M x N).
// M multiple of 256; N multiple of 64; K multiple of 32.
// ---------------------------------------------------------------------------
__global__ __launch_bounds__(128) void gemm_bf16_wmma_kernel(
    const unsigned short* __restrict__ A, const unsigned short* __restrict__ Bm,
    float* __restrict__ C, int M, int N, int K, int lda, int ldb, int ldc) {
  // A tile: 256 rows x 32 K bf16, stored as 16 K-pair dwords (+1 pad)
  __shared__ unsigned int AsU[256][17];
  // B tile stored transposed: 64 n-rows x 32 K as 16 K-pair dwords (+1 pad)
  __shared__ unsigned int BsU[64][17];
  unsigned short* Bss = (unsigned short*)&BsU[0][0];   // pitch 34 shorts per n-row

  const int tid  = threadIdx.x;
  const int lane = tid & 31;
  const int wave = tid >> 5;
  const int m0 = blockIdx.y * 256;
  const int n0 = blockIdx.x * 64;
  const int mW = wave * 64;            // wave's 64-row slice
  const int hi  = lane >> 4;
  const int l15 = lane & 15;

  v8f acc[4][4];                       // [m-subtile][n-subtile]
#pragma unroll
  for (int i = 0; i < 4; ++i)
#pragma unroll
    for (int j = 0; j < 4; ++j)
#pragma unroll
      for (int r = 0; r < 8; ++r) acc[i][j][r] = 0.0f;

  for (int k0 = 0; k0 < K; k0 += 32) {
    __syncthreads();   // previous compute done before LDS overwrite

    // ---- A tile: 256 rows x 32 bf16 = 1024 x 16B chunks, 8 per thread ----
#if USE_ASYNC_LDS
#pragma unroll
    for (int it = 0; it < 8; ++it) {
      int i = tid + it * 128;
      int row = i >> 2, c4 = i & 3;
      const unsigned short* gp = A + (size_t)(m0 + row) * lda + k0 + c4 * 8;
      __builtin_amdgcn_global_load_async_to_lds_b128(
          (v4i*)gp,
          (lds_v4i_ptr)&AsU[row][c4 * 4],
          0, 0);
    }
#else
#pragma unroll
    for (int it = 0; it < 8; ++it) {
      int i = tid + it * 128;
      int row = i >> 2, c4 = i & 3;
      const uint4 v = *(const uint4*)(A + (size_t)(m0 + row) * lda + k0 + c4 * 8);
      AsU[row][c4 * 4 + 0] = v.x;
      AsU[row][c4 * 4 + 1] = v.y;
      AsU[row][c4 * 4 + 2] = v.z;
      AsU[row][c4 * 4 + 3] = v.w;
    }
#endif

    // ---- B tile transposed: 32 k-rows x 64 bf16 = 256 uint4, 2/thread ----
#pragma unroll
    for (int it = 0; it < 2; ++it) {
      int i = tid + it * 128;
      int k = i >> 3, c = i & 7;
      const uint4 v = *(const uint4*)(Bm + (size_t)(k0 + k) * ldb + n0 + c * 8);
      const unsigned short* sv = (const unsigned short*)&v;
#pragma unroll
      for (int e = 0; e < 8; ++e) Bss[(size_t)(c * 8 + e) * 34 + k] = sv[e];
    }

    // prefetch next K tile (global_prefetch_b8)
    if (k0 + 32 < K) {
      __builtin_prefetch(A + (size_t)(m0 + (tid & 127) * 2) * lda + k0 + 32, 0, 1);
      __builtin_prefetch(Bm + (size_t)(k0 + 32 + (tid & 31)) * ldb + n0, 0, 1);
    }

#if USE_ASYNC_LDS
    __builtin_amdgcn_s_wait_asynccnt(0);   // my async LDS writes landed
#endif
    __syncthreads();

    // ---- fragments per ISA 7.12.2 layouts --------------------------------
    union FragU { unsigned int u[8]; v16bf v; };
    FragU bfrag[4];
#pragma unroll
    for (int j = 0; j < 4; ++j) {
      const int n = j * 16 + l15;
      // B 32x16 bf16: vgpr q holds K-pair (g*8 + hi*4 + r), g=q>>2, r=q&3, for column n
#pragma unroll
      for (int q = 0; q < 8; ++q) bfrag[j].u[q] = BsU[n][(q >> 2) * 8 + hi * 4 + (q & 3)];
    }
#pragma unroll
    for (int i2 = 0; i2 < 4; ++i2) {
      FragU af;
      const int m = mW + i2 * 16 + l15;
      // A 16x32 bf16: vgpr q holds K-pair (g*8 + hi*4 + r), g=q>>2, r=q&3
#pragma unroll
      for (int q = 0; q < 8; ++q) {
        int g = q >> 2, r = q & 3;
        af.u[q] = AsU[m][g * 8 + hi * 4 + r];
      }
#pragma unroll
      for (int j = 0; j < 4; ++j)
        acc[i2][j] = __builtin_amdgcn_wmma_f32_16x16x32_bf16(
            false, af.v, false, bfrag[j].v, (short)0, acc[i2][j], false, false);
    }
  }

  // ---- store: C/D f32 layout — vgpr r -> M = r + 8*hi, lane&15 -> N -------
#pragma unroll
  for (int i2 = 0; i2 < 4; ++i2)
#pragma unroll
    for (int j = 0; j < 4; ++j) {
      const int nn = n0 + j * 16 + l15;
#pragma unroll
      for (int r = 0; r < 8; ++r) {
        const int m = m0 + mW + i2 * 16 + r + 8 * hi;
        C[(size_t)m * ldc + nn] = acc[i2][j][r];
      }
    }
}

// ---------------------------------------------------------------------------
// Recurrent SSM scan. One 64-thread workgroup per (head, batch).
// Thread p keeps state row h[p][0..63] in 64 VGPRs. dt-cumsum (RoPE phase)
// carried as a running scalar. Produces y_pre (pre-gating) in (B,T,D_INNER).
// ---------------------------------------------------------------------------
__global__ __launch_bounds__(64) void ssm_scan_kernel(
    const float* __restrict__ zxbcdt, const float* __restrict__ dt_bias,
    const float* __restrict__ A_log, const float* __restrict__ D_param,
    const float* __restrict__ B_bias, const float* __restrict__ C_bias,
    const float* __restrict__ rope_freqs, float* __restrict__ ypre) {
  const int h = blockIdx.x;     // head
  const int b = blockIdx.y;     // batch
  const int n = threadIdx.x;    // doubles as state column n and row p

  __shared__ float BhL[64];
  __shared__ float ChL[64];
  __shared__ float red[4];

  float hstate[64];
#pragma unroll
  for (int i = 0; i < 64; ++i) hstate[i] = 0.0f;

  const float dtb   = dt_bias[h];
  const float Aexp  = expf(A_log[h]);
  const float Dp    = D_param[h];
  const float bbias = B_bias[h * 64 + n];
  const float cbias = C_bias[h * 64 + n];
  const float freq  = rope_freqs[h * 32 + (n >> 1)];
  float phi = 0.0f;

  const float* base = zxbcdt + (size_t)b * SEQ * D_PROJ;

  for (int t = 0; t < SEQ; ++t) {
    const float* row = base + (size_t)t * D_PROJ;
    const float braw = row[2 * D_INNER + n];                  // group 0 B
    const float craw = row[2 * D_INNER + D_STATE + n];        // group 0 C
    const float dtr  = row[2 * D_INNER + 2 * D_STATE + h];    // scalar broadcast
    const float xh   = row[D_INNER + h * HEAD_DIM + n];       // p = n

    // softplus(dt_raw + dt_bias)
    const float sp  = dtr + dtb;
    const float dtv = (sp > 20.0f) ? sp : log1pf(expf(sp));
    const float a   = expf(-Aexp * dtv);
    phi += dtv;

    // RMS sums of squares over 64 lanes (2 waves): wave shfl + LDS combine
    float sb = braw * braw, sc = craw * craw;
#pragma unroll
    for (int off = 16; off > 0; off >>= 1) {
      sb += __shfl_xor(sb, off, 32);
      sc += __shfl_xor(sc, off, 32);
    }
    const int wid = n >> 5;
    if ((n & 31) == 0) { red[wid * 2] = sb; red[wid * 2 + 1] = sc; }
    __syncthreads();
    sb = red[0] + red[2];
    sc = red[1] + red[3];

    float rb = braw * rsqrtf(sb * (1.0f / 64.0f) + EPS) + bbias;
    float rc = craw * rsqrtf(sc * (1.0f / 64.0f) + EPS) + cbias;

    // RoPE: pair (2i, 2i+1); partner via lane xor 1
    const float ang = phi * freq;
    const float cv = cosf(ang), sv = sinf(ang);
    const float rbp = __shfl_xor(rb, 1, 32);
    const float rcp = __shfl_xor(rc, 1, 32);
    float Bh, Ch;
    if ((n & 1) == 0) { Bh = rb * cv - rbp * sv; Ch = rc * cv - rcp * sv; }
    else              { Bh = rbp * sv + rb * cv; Ch = rcp * sv + rc * cv; }
    BhL[n] = Bh;
    ChL[n] = Ch;
    __syncthreads();

    // state update + output: h[p][j] = a*h + u_p*Bh[j];  y_p += h[p][j]*Ch[j]
    const float u = xh * dtv;
    float yp = 0.0f;
#pragma unroll
    for (int j = 0; j < 64; ++j) {
      hstate[j] = a * hstate[j] + u * BhL[j];
      yp += hstate[j] * ChL[j];
    }
    ypre[((size_t)b * SEQ + t) * D_INNER + h * HEAD_DIM + n] = SSD_SCALE * yp + Dp * xh;
    __syncthreads();   // protect red/BhL/ChL before next step
  }
}

// ---------------------------------------------------------------------------
// g = ypre * silu(z); RMSNorm(g) * norm_w -> bf16. One 256-thread block / row.
// ---------------------------------------------------------------------------
__global__ __launch_bounds__(256) void gate_norm_kernel(
    const float* __restrict__ zxbcdt, const float* __restrict__ ypre,
    const float* __restrict__ norm_w, unsigned short* __restrict__ gbf) {
  const int r   = blockIdx.x;          // (b*T + t)
  const int tid = threadIdx.x;
  const float* zrow = zxbcdt + (size_t)r * D_PROJ;   // z = first D_INNER cols
  const float* yrow = ypre   + (size_t)r * D_INNER;

  float g[16];
  float ss = 0.0f;
#pragma unroll
  for (int j = 0; j < 16; ++j) {
    const int i = tid + j * 256;
    const float z = zrow[i];
    const float sil = z / (1.0f + expf(-z));
    const float gv = yrow[i] * sil;
    g[j] = gv;
    ss += gv * gv;
  }
  __shared__ float redL[256];
  redL[tid] = ss;
  __syncthreads();
  for (int s = 128; s > 0; s >>= 1) {
    if (tid < s) redL[tid] += redL[tid + s];
    __syncthreads();
  }
  const float inv = rsqrtf(redL[0] * (1.0f / (float)D_INNER) + EPS);
#pragma unroll
  for (int j = 0; j < 16; ++j) {
    const int i = tid + j * 256;
    const float o = g[j] * inv * norm_w[i];
    unsigned int x = __float_as_uint(o);
    unsigned int rr = x + 0x7fffu + ((x >> 16) & 1u);
    gbf[(size_t)r * D_INNER + i] = (unsigned short)(rr >> 16);
  }
}

// ---------------------------------------------------------------------------
// Host-side orchestration
// ---------------------------------------------------------------------------
extern "C" void kernel_launch(void* const* d_in, const int* in_sizes, int n_in,
                              void* d_out, int out_size, void* d_ws, size_t ws_size,
                              hipStream_t stream) {
  const float* x       = (const float*)d_in[0];   // (B,T,D_MODEL)
  const float* W_in    = (const float*)d_in[1];   // (D_MODEL, D_PROJ)
  const float* W_out   = (const float*)d_in[2];   // (D_INNER, D_MODEL)
  const float* dt_bias = (const float*)d_in[3];
  const float* A_log   = (const float*)d_in[4];
  const float* D_param = (const float*)d_in[5];
  const float* B_bias  = (const float*)d_in[6];
  const float* C_bias  = (const float*)d_in[7];
  const float* rope_f  = (const float*)d_in[8];
  const float* norm_w  = (const float*)d_in[9];
  float* out = (float*)d_out;

  const size_t n_x  = (size_t)BATCH * SEQ * D_MODEL;       // 8,388,608
  const size_t n_wi = (size_t)D_MODEL * D_PROJ;            // 17,170,432
  const size_t n_wo = (size_t)D_INNER * D_MODEL;           // 8,388,608
  const size_t n_zx = (size_t)BATCH * SEQ * D_PROJ;        // 34,340,864
  const size_t n_yp = (size_t)BATCH * SEQ * D_INNER;       // 16,777,216

  char* ws = (char*)d_ws;
  size_t off = 0;
  unsigned short* xbf  = (unsigned short*)(ws + off); off += n_x  * 2;
  unsigned short* wbf  = (unsigned short*)(ws + off); off += n_wi * 2;
  unsigned short* wobf = (unsigned short*)(ws + off); off += n_wo * 2;
  float*          zx   = (float*)(ws + off);          off += n_zx * 4;
  float*          ypre = (float*)(ws + off);          off += n_yp * 4;
  unsigned short* gbf  = (unsigned short*)(ws + off); off += n_yp * 2;
  (void)ws_size; (void)in_sizes; (void)n_in; (void)out_size;

  // 1) bf16 conversions
  f32_to_bf16_kernel<<<4096, 256, 0, stream>>>(x,     xbf,  n_x);
  f32_to_bf16_kernel<<<8192, 256, 0, stream>>>(W_in,  wbf,  n_wi);
  f32_to_bf16_kernel<<<4096, 256, 0, stream>>>(W_out, wobf, n_wo);

  // 2) in-projection: zxbcdt = x @ W_in   (M=4096, N=8384, K=2048)
  gemm_bf16_wmma_kernel<<<dim3(D_PROJ / 64, (BATCH * SEQ) / 256), 128, 0, stream>>>(
      xbf, wbf, zx, BATCH * SEQ, D_PROJ, D_MODEL, D_MODEL, D_PROJ, D_PROJ);

  // 3) recurrent SSM scan -> ypre
  ssm_scan_kernel<<<dim3(N_HEADS, BATCH), 64, 0, stream>>>(
      zx, dt_bias, A_log, D_param, B_bias, C_bias, rope_f, ypre);

  // 4) SiLU gate + RMSNorm -> bf16
  gate_norm_kernel<<<BATCH * SEQ, 256, 0, stream>>>(zx, ypre, norm_w, gbf);

  // 5) out-projection: out = g @ W_out   (M=4096, N=2048, K=4096)
  gemm_bf16_wmma_kernel<<<dim3(D_MODEL / 64, (BATCH * SEQ) / 256), 128, 0, stream>>>(
      gbf, wobf, out, BATCH * SEQ, D_MODEL, D_INNER, D_INNER, D_MODEL, D_MODEL);
}
